// PosMLPAttention_14061722927241
// MI455X (gfx1250) — matrix operand order repaired and checked by reading, not compiled, not run on hardware
//
#include <hip/hip_runtime.h>
#include <hip/hip_bf16.h>

typedef __attribute__((ext_vector_type(16))) _Float16 v16h;
typedef __attribute__((ext_vector_type(8)))  float    v8f;

#define NB      128
#define DIMC    256
#define NHEADS  8
#define HEADD   32
#define HIDDEN  16
#define NKEYS   4096      // 64*64
#define GRIDW   64
#define LDSPITCH 257      // padded row stride (floats): 16*257 % 64 == 16 -> no bank conflicts

// ---------------------------------------------------------------------------
// Kernel 1: per-batch prep.
//   Q[b,j]      = queries[b,:] . q_w[j,:]
//   qproj[b,h,c]= scale * sum_d Q[b,h*32+d] * kv_w[h*32+d, c]   (K half of kv_w)
//   wparams[b,j]= queries[b,:] . wg_w[j,:] + wg_b[j]            (j < 65)
// ---------------------------------------------------------------------------
__global__ void prep_kernel(const float* __restrict__ queries,
                            const float* __restrict__ q_w,
                            const float* __restrict__ kv_w,
                            const float* __restrict__ wg_w,
                            const float* __restrict__ wg_b,
                            float* __restrict__ qproj,
                            float* __restrict__ wparams) {
  const int b   = blockIdx.x;
  const int tid = threadIdx.x;
  __shared__ float qin[DIMC];
  __shared__ float Qs[DIMC];

  qin[tid] = queries[(size_t)b * DIMC + tid];
  __syncthreads();

  {
    const float* wr = q_w + (size_t)tid * DIMC;
    float s = 0.f;
    for (int k = 0; k < DIMC; ++k) s += qin[k] * wr[k];
    Qs[tid] = s;
  }
  if (tid < 4 * HIDDEN + 1) {
    const float* wr = wg_w + (size_t)tid * DIMC;
    float s = 0.f;
    for (int k = 0; k < DIMC; ++k) s += qin[k] * wr[k];
    wparams[(size_t)b * 65 + tid] = s + wg_b[tid];
  }
  __syncthreads();

  const float scale = 0.17677669529663689f;   // 32^-0.5
  const int c = tid;
  for (int h = 0; h < NHEADS; ++h) {
    float s = 0.f;
    for (int d = 0; d < HEADD; ++d)
      s += Qs[h * HEADD + d] * kv_w[(size_t)(h * HEADD + d) * DIMC + c];
    qproj[((size_t)b * NHEADS + h) * DIMC + c] = s * scale;
  }
}

// ---------------------------------------------------------------------------
// Kernel 2: box bias per (b, n).  grid (B,16) x 256 threads.
// ---------------------------------------------------------------------------
__global__ void bias_kernel(const float* __restrict__ pos,
                            const float* __restrict__ wparams,
                            float* __restrict__ biasb) {
  const int b = blockIdx.x;
  const int n = blockIdx.y * 256 + threadIdx.x;
  __shared__ float wp[65];
  __shared__ float p4[4];
  if (threadIdx.x < 65) wp[threadIdx.x] = wparams[(size_t)b * 65 + threadIdx.x];
  if (threadIdx.x < 4)  p4[threadIdx.x] = pos[(size_t)b * 4 + threadIdx.x];
  __syncthreads();

  const int x = n & (GRIDW - 1);
  const int y = n >> 6;
  const float gx = (x + 0.5f) / (float)GRIDW;
  const float gy = (y + 0.5f) / (float)GRIDW;
  const float dx = (gx - p4[0]) / p4[2];
  const float dy = (gy - p4[1]) / p4[3];
  float acc = wp[64];
  #pragma unroll
  for (int k = 0; k < HIDDEN; ++k) {
    float hk = dx * wp[2 * k] + dy * wp[2 * k + 1] + wp[2 * HIDDEN + k];
    hk = fmaxf(hk, 0.f);
    acc += hk * wp[3 * HIDDEN + k];
  }
  biasb[(size_t)b * NKEYS + n] = acc;
}

// ---------------------------------------------------------------------------
// Kernel 3: scores[b,h,n] = mem[b,n,:].qproj[b,h,:] + bias[b,n]  via WMMA.
// grid (B, 8): each block = 512 keys, 8 waves x 4 row-tiles of 16 keys.
// A = mem tile [16 x 256] f16 (b128 row loads), B = qproj^T in 8 v16h regs.
// ---------------------------------------------------------------------------
__global__ void scores_kernel(const float* __restrict__ mem,
                              const float* __restrict__ qproj,
                              const float* __restrict__ biasb,
                              float* __restrict__ scores) {
  const int b      = blockIdx.x;
  const int chunk  = blockIdx.y;
  const int wave   = threadIdx.x >> 5;
  const int lane   = threadIdx.x & 31;
  const int laneHi = lane >> 4;
  const int lanelo = lane & 15;
  const int head   = lanelo;

  v16h Bfrag[8];
  {
    const float* qp = qproj + (size_t)b * NHEADS * DIMC;
    #pragma unroll
    for (int kk = 0; kk < 8; ++kk) {
      v16h t;
      #pragma unroll
      for (int e = 0; e < 16; ++e) {
        const int k = kk * 32 + (e & 7) + ((e >> 3) << 4) + (laneHi ? 8 : 0);
        const float v = (head < NHEADS) ? qp[(size_t)head * DIMC + k] : 0.f;
        t[e] = (_Float16)v;
      }
      Bfrag[kk] = t;
    }
  }

  const float* memb = mem + (size_t)b * NKEYS * DIMC;
  for (int t = 0; t < 4; ++t) {
    const int n0 = chunk * 512 + (wave * 4 + t) * 16;
    const float* row = memb + (size_t)(n0 + lanelo) * DIMC;
    __builtin_prefetch(row + 16 * DIMC, 0, 0);   // next tile, same lane row
    v8f acc = {};
    #pragma unroll
    for (int kk = 0; kk < 8; ++kk) {
      const int kb = kk * 32 + (laneHi ? 8 : 0);
      const float4 f0 = *reinterpret_cast<const float4*>(row + kb);
      const float4 f1 = *reinterpret_cast<const float4*>(row + kb + 4);
      const float4 f2 = *reinterpret_cast<const float4*>(row + kb + 16);
      const float4 f3 = *reinterpret_cast<const float4*>(row + kb + 20);
      v16h a;
      a[0]=(_Float16)f0.x; a[1]=(_Float16)f0.y; a[2]=(_Float16)f0.z; a[3]=(_Float16)f0.w;
      a[4]=(_Float16)f1.x; a[5]=(_Float16)f1.y; a[6]=(_Float16)f1.z; a[7]=(_Float16)f1.w;
      a[8]=(_Float16)f2.x; a[9]=(_Float16)f2.y; a[10]=(_Float16)f2.z; a[11]=(_Float16)f2.w;
      a[12]=(_Float16)f3.x; a[13]=(_Float16)f3.y; a[14]=(_Float16)f3.z; a[15]=(_Float16)f3.w;
      acc = __builtin_amdgcn_wmma_f32_16x16x32_f16(false, a, false, Bfrag[kk],
                                                   (short)0, acc, false, false);
    }
    if (head < NHEADS) {
      float* sc = scores + ((size_t)b * NHEADS + head) * NKEYS;
      const float* bb = biasb + (size_t)b * NKEYS;
      #pragma unroll
      for (int r = 0; r < 8; ++r) {
        const int n = n0 + r + laneHi * 8;
        sc[n] = acc[r] + bb[n];
      }
    }
  }
}

// ---------------------------------------------------------------------------
// Kernel 4: softmax over n per (b,h) row. Stores unnormalized exp in place
// and the row sum in l[row]. grid = B*8 blocks x 256 threads.
// ---------------------------------------------------------------------------
__global__ void softmax_kernel(float* __restrict__ scores,
                               float* __restrict__ lbuf) {
  const int row = blockIdx.x;
  const int tid = threadIdx.x;
  float* s = scores + (size_t)row * NKEYS;
  __shared__ float red[256];

  float m = -3.4e38f;
  #pragma unroll
  for (int i = 0; i < 16; ++i) m = fmaxf(m, s[tid + i * 256]);
  red[tid] = m;
  __syncthreads();
  for (int d = 128; d > 0; d >>= 1) {
    if (tid < d) red[tid] = fmaxf(red[tid], red[tid + d]);
    __syncthreads();
  }
  m = red[0];
  __syncthreads();

  float sum = 0.f;
  #pragma unroll
  for (int i = 0; i < 16; ++i) {
    const float e = __expf(s[tid + i * 256] - m);
    s[tid + i * 256] = e;
    sum += e;
  }
  red[tid] = sum;
  __syncthreads();
  for (int d = 128; d > 0; d >>= 1) {
    if (tid < d) red[tid] += red[tid + d];
    __syncthreads();
  }
  if (tid == 0) lbuf[row] = red[0];
}

// ---------------------------------------------------------------------------
// Kernel 5: pm_part[b,slice,h,c] = sum_{n in slice} p[b,h,n] * mem[b,n,c].
// grid (B,4): slice = 1024 keys = 32 chunks of 32 keys (one WMMA k-step each).
// Each chunk (32 x 256 fp32 = 32 KB) is staged into padded LDS with
// global_load_async_to_lds_b128 (ASYNCcnt), then 8 waves build transposed
// B fragments from LDS (conflict-free via 257-float pitch) and run 2 WMMAs
// each (2 column tiles of 16 channels per wave).
// A = P [16(head) x 32(n)] f16 (8 heads valid), b128 loads from p buffer.
// ---------------------------------------------------------------------------
__global__ void pm_kernel(const float* __restrict__ mem,
                          const float* __restrict__ pbuf,
                          float* __restrict__ pm_part) {
  const int b      = blockIdx.x;
  const int slice  = blockIdx.y;
  const int tid    = threadIdx.x;
  const int wave   = tid >> 5;
  const int lane   = tid & 31;
  const int laneHi = lane >> 4;
  const int lanelo = lane & 15;
  const bool pvalid = (lanelo < NHEADS);

  __shared__ float buf[32 * LDSPITCH];   // 32.9 KB staged chunk, padded rows

  const float* memb = mem + (size_t)b * NKEYS * DIMC;
  const float* prow = pbuf + ((size_t)b * NHEADS + (pvalid ? lanelo : 0)) * NKEYS;

  v8f acc0 = {}, acc1 = {};
  const int c0a = (wave * 2 + 0) * 16;
  const int c0b = (wave * 2 + 1) * 16;

  for (int ks = 0; ks < 32; ++ks) {
    const int n0 = slice * 1024 + ks * 32;

    // ---- async stage: 32 rows x 1 KB, 16 B per lane-transfer, 8 per thread
    #pragma unroll
    for (int i = 0; i < 8; ++i) {
      const int linear = i * 256 + tid;        // 0..2047
      const int row    = linear >> 6;          // 0..31
      const int c16    = linear & 63;          // 16B chunk within row
      const unsigned ldsoff = (unsigned)(size_t)&buf[row * LDSPITCH + c16 * 4];
      const float* g = memb + (size_t)(n0 + row) * DIMC + c16 * 4;
      asm volatile("global_load_async_to_lds_b128 %0, %1, off"
                   :: "v"(ldsoff), "v"(g) : "memory");
    }
    asm volatile("s_wait_asynccnt 0x0" ::: "memory");
    __syncthreads();

    // ---- A fragment: row = lanelo (head), K = n per ISA A-layout (b128 loads)
    v16h a;
    {
      float4 p0, p1, p2, p3;
      if (pvalid) {
        p0 = *reinterpret_cast<const float4*>(prow + n0 + laneHi * 8);
        p1 = *reinterpret_cast<const float4*>(prow + n0 + laneHi * 8 + 4);
        p2 = *reinterpret_cast<const float4*>(prow + n0 + 16 + laneHi * 8);
        p3 = *reinterpret_cast<const float4*>(prow + n0 + 16 + laneHi * 8 + 4);
      } else {
        p0 = p1 = p2 = p3 = make_float4(0.f, 0.f, 0.f, 0.f);
      }
      a[0]=(_Float16)p0.x; a[1]=(_Float16)p0.y; a[2]=(_Float16)p0.z; a[3]=(_Float16)p0.w;
      a[4]=(_Float16)p1.x; a[5]=(_Float16)p1.y; a[6]=(_Float16)p1.z; a[7]=(_Float16)p1.w;
      a[8]=(_Float16)p2.x; a[9]=(_Float16)p2.y; a[10]=(_Float16)p2.z; a[11]=(_Float16)p2.w;
      a[12]=(_Float16)p3.x; a[13]=(_Float16)p3.y; a[14]=(_Float16)p3.z; a[15]=(_Float16)p3.w;
    }

    // ---- B fragments from LDS: col = lanelo, k = (e&7) + (e>>3)*16 + laneHi*8
    v16h b0, b1;
    #pragma unroll
    for (int e = 0; e < 16; ++e) {
      const int rowoff = ((e & 7) + ((e >> 3) << 4) + laneHi * 8) * LDSPITCH;
      b0[e] = (_Float16)buf[rowoff + c0a + lanelo];
      b1[e] = (_Float16)buf[rowoff + c0b + lanelo];
    }

    acc0 = __builtin_amdgcn_wmma_f32_16x16x32_f16(false, a, false, b0,
                                                  (short)0, acc0, false, false);
    acc1 = __builtin_amdgcn_wmma_f32_16x16x32_f16(false, a, false, b1,
                                                  (short)0, acc1, false, false);
    __syncthreads();   // protect buf before next chunk's async writes
  }

  // D: col = lanelo + c0, row(head) = r + laneHi*8 -> only laneHi==0 valid
  if (laneHi == 0) {
    float* dst = pm_part + ((size_t)b * 4 + slice) * (NHEADS * DIMC);
    #pragma unroll
    for (int r = 0; r < 8; ++r) {
      dst[(size_t)r * DIMC + c0a + lanelo] = acc0[r];
      dst[(size_t)r * DIMC + c0b + lanelo] = acc1[r];
    }
  }
}

// ---------------------------------------------------------------------------
// Kernel 6: reduce partials, attnV = (pm . Wv) / l, out = attnV @ out_w^T.
// ---------------------------------------------------------------------------
__global__ void final_kernel(const float* __restrict__ pm_part,
                             const float* __restrict__ lbuf,
                             const float* __restrict__ kv_w,
                             const float* __restrict__ out_w,
                             float* __restrict__ out) {
  const int b   = blockIdx.x;
  const int tid = threadIdx.x;
  __shared__ float pmsum[NHEADS * DIMC];
  __shared__ float att[DIMC];

  for (int i = tid; i < NHEADS * DIMC; i += 256) {
    float s = 0.f;
    #pragma unroll
    for (int sl = 0; sl < 4; ++sl)
      s += pm_part[((size_t)b * 4 + sl) * (NHEADS * DIMC) + i];
    pmsum[i] = s;
  }
  __syncthreads();

  const int h = tid >> 5, d = tid & 31;
  const float* wv = kv_w + (size_t)(DIMC + h * HEADD + d) * DIMC;  // V half
  float a = 0.f;
  for (int c = 0; c < DIMC; ++c) a += pmsum[h * DIMC + c] * wv[c];
  att[tid] = a / lbuf[(size_t)b * NHEADS + h];
  __syncthreads();

  const float* ow = out_w + (size_t)tid * DIMC;
  float o = 0.f;
  for (int k = 0; k < DIMC; ++k) o += att[k] * ow[k];
  out[(size_t)b * DIMC + tid] = o;
}

// ---------------------------------------------------------------------------
extern "C" void kernel_launch(void* const* d_in, const int* in_sizes, int n_in,
                              void* d_out, int out_size, void* d_ws, size_t ws_size,
                              hipStream_t stream) {
  const float* queries = (const float*)d_in[0];  // [128,256]
  const float* memory  = (const float*)d_in[1];  // [128,64,64,256]
  const float* pos     = (const float*)d_in[2];  // [128,4]
  const float* q_w     = (const float*)d_in[3];  // [256,256]
  const float* kv_w    = (const float*)d_in[4];  // [512,256]
  const float* out_w   = (const float*)d_in[5];  // [256,256]
  const float* wg_w    = (const float*)d_in[6];  // [65,256]
  const float* wg_b    = (const float*)d_in[7];  // [65]
  float* out = (float*)d_out;

  // Workspace layout (floats)
  float* ws      = (float*)d_ws;
  float* qproj   = ws;                                    // 128*8*256   = 262144
  float* wparams = qproj + (size_t)NB * NHEADS * DIMC;    // 128*65      = 8320
  float* biasb   = wparams + (size_t)NB * 65;             // 128*4096    = 524288
  float* scores  = biasb + (size_t)NB * NKEYS;            // 128*8*4096  = 4194304
  float* lbuf    = scores + (size_t)NB * NHEADS * NKEYS;  // 128*8       = 1024
  float* pm_part = lbuf + (size_t)NB * NHEADS;            // 128*4*8*256 = 1048576

  prep_kernel<<<NB, 256, 0, stream>>>(queries, q_w, kv_w, wg_w, wg_b, qproj, wparams);
  bias_kernel<<<dim3(NB, 16), 256, 0, stream>>>(pos, wparams, biasb);
  scores_kernel<<<dim3(NB, 8), 256, 0, stream>>>(memory, qproj, biasb, scores);
  softmax_kernel<<<NB * NHEADS, 256, 0, stream>>>(scores, lbuf);
  pm_kernel<<<dim3(NB, 4), 256, 0, stream>>>(memory, scores, pm_part);
  final_kernel<<<NB, 256, 0, stream>>>(pm_part, lbuf, kv_w, out_w, out);
}
